// LocalEmbedder_22428319220593
// MI455X (gfx1250) — matrix operand order repaired and checked by reading, not compiled, not run on hardware
//
#include <hip/hip_runtime.h>
#include <math.h>

// ---------------------------------------------------------------------------
// DGCNN LocalEmbedder for MI455X (gfx1250, wave32, WMMA, 320KB LDS)
// B=8, C=3, N=4096, D=64, k=20
// ---------------------------------------------------------------------------

#define BB 8
#define NN 4096
#define KK 20
#define DD 64
#define CNT1 ((float)(BB * NN * KK))

typedef __attribute__((ext_vector_type(16))) __bf16 v16bf;
typedef __attribute__((ext_vector_type(8)))  float  v8f;

union V16U { v16bf v; unsigned u[8]; };

// ---------------------------------------------------------------------------
// Top-20 selection on one LDS row of 4096 f32, one wave32 per row.
// 20 rounds of (lane-strided scan -> full-butterfly wave argmax -> knock out).
// All lanes converge on the same argmax via xor butterfly; descending order
// matches jax.lax.top_k.
// ---------------------------------------------------------------------------
__device__ __forceinline__ void topk20_row(float* row, int* out_idx, int lane) {
  for (int it = 0; it < KK; ++it) {
    float best = -INFINITY; int bidx = 0x7fffffff;
    for (int j = lane; j < NN; j += 32) {
      float v = row[j];
      if (v > best) { best = v; bidx = j; }
    }
    #pragma unroll
    for (int off = 16; off > 0; off >>= 1) {
      float ob = __shfl_xor(best, off);
      int   oi = __shfl_xor(bidx, off);
      if (ob > best || (ob == best && oi < bidx)) { best = ob; bidx = oi; }
    }
    if (lane == 0) out_idx[it] = bidx;
    if ((bidx & 31) == lane) row[bidx] = -INFINITY;  // knock out for next round
  }
}

// ---------------------------------------------------------------------------
// K0: zero the 512-float stats block (sums/scales/shifts)
// ---------------------------------------------------------------------------
__global__ void k_init_stats(float* stats) {
  int i = threadIdx.x;
  if (i < 512) stats[i] = 0.0f;
}

// ---------------------------------------------------------------------------
// K1: xx1[b,n] = sum_c x[b,c,n]^2   (C=3)
// ---------------------------------------------------------------------------
__global__ void k_xx1(const float* __restrict__ x, float* __restrict__ xx1) {
  int i = blockIdx.x * blockDim.x + threadIdx.x;      // (b,n)
  if (i >= BB * NN) return;
  int b = i >> 12, n = i & (NN - 1);
  const float* xb = x + (size_t)b * 3 * NN;
  float s = 0.f;
  #pragma unroll
  for (int c = 0; c < 3; ++c) { float v = xb[c * NN + n]; s += v * v; }
  xx1[i] = s;
}

// ---------------------------------------------------------------------------
// K2: KNN on raw 3-dim points. 4 rows per 128-thread block, 64KB LDS strip.
// Stored score = 2*dot - xx[m]  (the -xx[n] row constant cannot change order)
// ---------------------------------------------------------------------------
__global__ __launch_bounds__(128) void k_knn1(const float* __restrict__ x,
                                              const float* __restrict__ xx1,
                                              int* __restrict__ idx1) {
  __shared__ float dist[4][NN];                       // 64 KB
  int b  = blockIdx.x >> 10;                          // 1024 row-blocks / batch
  int n0 = (blockIdx.x & 1023) * 4;
  const float* xb = x + (size_t)b * 3 * NN;
  for (int t = threadIdx.x; t < 4 * NN; t += 128) {
    int r = t >> 12, m = t & (NN - 1);
    int n = n0 + r;
    float d = 0.f;
    #pragma unroll
    for (int c = 0; c < 3; ++c) d += xb[c * NN + n] * xb[c * NN + m];
    dist[r][m] = 2.f * d - xx1[b * NN + m];
  }
  __syncthreads();
  int wave = threadIdx.x >> 5, lane = threadIdx.x & 31;
  topk20_row(&dist[wave][0], idx1 + (size_t)(b * NN + n0 + wave) * KK, lane);
}

// ---------------------------------------------------------------------------
// K3: EdgeConv1 (64x6 conv) fused with BN statistics and per-(b,o,n) max/min
// over k.  4 points per 256-thread block; thread = (point, out-channel).
// ---------------------------------------------------------------------------
__global__ __launch_bounds__(256) void k_conv1(const float* __restrict__ x,
                                               const int* __restrict__ idx1,
                                               const float* __restrict__ W1,
                                               float* __restrict__ ymax1,
                                               float* __restrict__ ymin1,
                                               float* __restrict__ sum1,
                                               float* __restrict__ sumsq1) {
  __shared__ float W1s[DD * 6];
  __shared__ float red[256];
  int b  = blockIdx.x >> 10;
  int n0 = (blockIdx.x & 1023) * 4;
  for (int t = threadIdx.x; t < DD * 6; t += 256) W1s[t] = W1[t];
  __syncthreads();
  int pl = threadIdx.x >> 6;                          // point 0..3
  int o  = threadIdx.x & 63;                          // output channel
  int n  = n0 + pl;
  const float* xb = x + (size_t)b * 3 * NN;
  float xi[3];
  #pragma unroll
  for (int c = 0; c < 3; ++c) xi[c] = xb[c * NN + n];
  float mx = -INFINITY, mn = INFINITY, s = 0.f, ss = 0.f;
  for (int k = 0; k < KK; ++k) {
    int j = idx1[(size_t)(b * NN + n) * KK + k];
    float y = 0.f;
    #pragma unroll
    for (int c = 0; c < 3; ++c) {
      y += W1s[o * 6 + c]     * (xb[c * NN + j] - xi[c]);  // edge feature
      y += W1s[o * 6 + 3 + c] * xi[c];                     // center feature
    }
    mx = fmaxf(mx, y); mn = fminf(mn, y); s += y; ss += y * y;
  }
  size_t oi = ((size_t)b * DD + o) * NN + n;
  ymax1[oi] = mx; ymin1[oi] = mn;
  red[threadIdx.x] = s;  __syncthreads();
  if (pl == 0) atomicAdd(&sum1[o],   red[o] + red[64 + o] + red[128 + o] + red[192 + o]);
  __syncthreads();
  red[threadIdx.x] = ss; __syncthreads();
  if (pl == 0) atomicAdd(&sumsq1[o], red[o] + red[64 + o] + red[128 + o] + red[192 + o]);
}

// ---------------------------------------------------------------------------
// K4: BN finalize: scale = g*rsqrt(var+eps), shift = beta - mean*scale
// ---------------------------------------------------------------------------
__global__ void k_finalize(const float* __restrict__ sum, const float* __restrict__ sumsq,
                           const float* __restrict__ g, const float* __restrict__ beta,
                           float* __restrict__ scale, float* __restrict__ shift, float count) {
  int o = threadIdx.x;
  if (o >= DD) return;
  float mean = sum[o] / count;
  float var  = sumsq[o] / count - mean * mean;
  float sc   = g[o] * rsqrtf(var + 1e-5f);
  scale[o] = sc;
  shift[o] = beta[o] - mean * sc;
}

// ---------------------------------------------------------------------------
// K5: activation: x1 = lrelu(scale*sel + shift), sel = scale>=0 ? ymax : ymin
// (lrelu monotone increasing => max over k commutes through the affine map).
// Also emits bf16 copy for WMMA.
// ---------------------------------------------------------------------------
__global__ void k_act1(const float* __restrict__ ymax1, const float* __restrict__ ymin1,
                       const float* __restrict__ scale1, const float* __restrict__ shift1,
                       float* __restrict__ x1, __bf16* __restrict__ x1h) {
  size_t i = (size_t)blockIdx.x * blockDim.x + threadIdx.x;
  if (i >= (size_t)BB * DD * NN) return;
  int o = (int)((i >> 12) & 63);
  float sc = scale1[o];
  float v  = sc * (sc >= 0.f ? ymax1[i] : ymin1[i]) + shift1[o];
  v = (v >= 0.f) ? v : 0.2f * v;
  x1[i]  = v;
  x1h[i] = (__bf16)v;
}

// ---------------------------------------------------------------------------
// K6: xx2[b,n] = sum_o x1[b,o,n]^2
// ---------------------------------------------------------------------------
__global__ void k_xx2(const float* __restrict__ x1, float* __restrict__ xx2) {
  int i = blockIdx.x * blockDim.x + threadIdx.x;
  if (i >= BB * NN) return;
  int b = i >> 12, n = i & (NN - 1);
  const float* xb = x1 + (size_t)b * DD * NN;
  float s = 0.f;
  for (int o = 0; o < DD; ++o) { float v = xb[o * NN + n]; s += v * v; }
  xx2[i] = s;
}

// ---------------------------------------------------------------------------
// K7: KNN on 64-dim features via bf16 WMMA.
// One workgroup (512 threads = 16 wave32) owns a 16-row strip:
//   - A tile (16 rows x 64 ch) staged once in LDS, K-major rows for 32-bit
//     fragment reads.
//   - Each wave computes 16 disjoint 16x16 Gram tiles with
//     v_wmma_f32_16x16x32_bf16 (K=64 -> 2 steps), writes scores straight
//     into a 16x4096 f32 LDS strip (256 KB — CDNA5's 320KB LDS).
//   - Then each wave runs top-20 on its own row, all in LDS.
// Score = 2*S[n,m] - xx2[m]  (row-constant -xx2[n] dropped; order-preserving)
// ---------------------------------------------------------------------------
__global__ __launch_bounds__(512) void k_knn2(const __bf16* __restrict__ x1h,
                                              const float* __restrict__ xx2,
                                              int* __restrict__ idx2) {
  __shared__ float  dist[16][NN];                     // 256 KB
  __shared__ __bf16 Als[16][DD];                      //   2 KB  [row M][c]
  __shared__ __bf16 Bls[16][16][DD];                  //  32 KB  [wave][col N][c]

  int b  = blockIdx.x >> 8;                           // 256 strips / batch
  int n0 = (blockIdx.x & 255) * 16;
  const __bf16* xb = x1h + (size_t)b * DD * NN;

  for (int t = threadIdx.x; t < 16 * DD; t += 512) {  // stage A tile
    int M = t >> 6, c = t & 63;
    Als[M][c] = xb[c * NN + n0 + M];
  }
  __syncthreads();

  int wave = threadIdx.x >> 5, lane = threadIdx.x & 31;
  int hi = lane >> 4, lo = lane & 15;

  for (int i = 0; i < 16; ++i) {
    int m0 = (wave + 16 * i) * 16;                    // disjoint col tiles
    for (int e = lane; e < 16 * DD; e += 32) {        // stage B tile (wave-private)
      int Nl = e >> 6, c = e & 63;
      Bls[wave][Nl][c] = xb[c * NN + m0 + Nl];
    }
    if (i < 15) {                                     // gfx1250 global_prefetch_b8
      int mnext = (wave + 16 * (i + 1)) * 16;
      __builtin_prefetch(&xb[mnext + lane], 0, 1);
    }
    v8f acc = {0.f, 0.f, 0.f, 0.f, 0.f, 0.f, 0.f, 0.f};
    #pragma unroll
    for (int kk = 0; kk < 2; ++kk) {
      V16U a, bm;
      #pragma unroll
      for (int v = 0; v < 8; ++v) {
        int ca = 32 * kk + ((v < 4) ? (2 * v + 8 * hi) : (16 + 2 * (v - 4) + 8 * hi));
        a.u[v]  = ((const unsigned*)&Als[lo][0])[ca >> 1];
        int cb = 32 * kk + ((v < 4) ? (2 * v + 8 * hi) : (16 + 2 * (v - 4) + 8 * hi));
        bm.u[v] = ((const unsigned*)&Bls[wave][lo][0])[cb >> 1];
      }
      acc = __builtin_amdgcn_wmma_f32_16x16x32_bf16(
          false, a.v, false, bm.v, (short)0, acc, false, false);
    }
    float xxm = xx2[b * NN + m0 + lo];
    #pragma unroll
    for (int v = 0; v < 8; ++v) {
      int r = v + 8 * hi;                             // D layout: row v+8*hi, col lo
      dist[r][m0 + lo] = 2.f * acc[v] - xxm;
    }
  }
  __syncthreads();
  topk20_row(&dist[wave][0], idx2 + (size_t)(b * NN + n0 + wave) * KK, lane);
}

// ---------------------------------------------------------------------------
// K8: EdgeConv2 (64x128 conv) via bf16 WMMA, fused with BN stats and
// per-(b,o,n) max/min over k.  One 256-thread block per (b,n):
//   feat2 (128 x 20, padded to 32 cols) in LDS; W2 in LDS as bf16;
//   8 waves = 4 M-tiles x 2 N-tiles, 4 K-steps each (K=128).
// Padding cols masked out of max/min/sum with +-inf / 0.
// ---------------------------------------------------------------------------
__global__ __launch_bounds__(256) void k_conv2(const float* __restrict__ x1,
                                               const float* __restrict__ W2,
                                               const int* __restrict__ idx2,
                                               float* __restrict__ ymax2,
                                               float* __restrict__ ymin2,
                                               float* __restrict__ sum2,
                                               float* __restrict__ sumsq2) {
  __shared__ __bf16 W2s[DD][128];                     // 16 KB [o][c]
  __shared__ __bf16 F2[32][128];                      //  8 KB [col][c]
  __shared__ float pmax[DD][2], pmin[DD][2], psum[DD][2], pss[DD][2];

  int b = blockIdx.x >> 12;
  int n = blockIdx.x & (NN - 1);
  const float* xb = x1 + (size_t)b * DD * NN;

  for (int t = threadIdx.x; t < DD * 128; t += 256)
    W2s[t >> 7][t & 127] = (__bf16)W2[t];
  for (int t = threadIdx.x; t < 32 * 128; t += 256) {
    int col = t >> 7, c = t & 127;
    float v = 0.f;
    if (col < KK) {
      int j = idx2[(size_t)(b * NN + n) * KK + col];
      v = (c < DD) ? (xb[c * NN + j] - xb[c * NN + n]) : xb[(c - DD) * NN + n];
    }
    F2[col][c] = (__bf16)v;
  }
  __syncthreads();

  int wave = threadIdx.x >> 5, lane = threadIdx.x & 31;
  int hi = lane >> 4, lo = lane & 15;
  int Mt = wave & 3, Nt = wave >> 2;

  v8f acc = {0.f, 0.f, 0.f, 0.f, 0.f, 0.f, 0.f, 0.f};
  #pragma unroll
  for (int kk = 0; kk < 4; ++kk) {
    V16U a, bm;
    #pragma unroll
    for (int v = 0; v < 8; ++v) {
      int ca = 32 * kk + ((v < 4) ? (2 * v + 8 * hi) : (16 + 2 * (v - 4) + 8 * hi));
      a.u[v]  = ((const unsigned*)&W2s[16 * Mt + lo][0])[ca >> 1];
      int cb = 32 * kk + ((v < 4) ? (2 * v + 8 * hi) : (16 + 2 * (v - 4) + 8 * hi));
      bm.u[v] = ((const unsigned*)&F2[16 * Nt + lo][0])[cb >> 1];
    }
    acc = __builtin_amdgcn_wmma_f32_16x16x32_bf16(
        false, a.v, false, bm.v, (short)0, acc, false, false);
  }

  int col = 16 * Nt + lo;
  bool valid = col < KK;
  #pragma unroll
  for (int v = 0; v < 8; ++v) {
    float y  = acc[v];
    float mx = valid ? y : -INFINITY;
    float mn = valid ? y :  INFINITY;
    float s  = valid ? y : 0.f;
    float ss = valid ? y * y : 0.f;
    #pragma unroll
    for (int off = 1; off < 16; off <<= 1) {          // reduce within 16-lane half
      mx = fmaxf(mx, __shfl_xor(mx, off));
      mn = fminf(mn, __shfl_xor(mn, off));
      s  += __shfl_xor(s,  off);
      ss += __shfl_xor(ss, off);
    }
    if (lo == 0) {
      int o = 16 * Mt + v + 8 * hi;
      pmax[o][Nt] = mx; pmin[o][Nt] = mn; psum[o][Nt] = s; pss[o][Nt] = ss;
    }
  }
  __syncthreads();
  if (threadIdx.x < DD) {
    int o = threadIdx.x;
    size_t oi = ((size_t)b * DD + o) * NN + n;
    ymax2[oi] = fmaxf(pmax[o][0], pmax[o][1]);
    ymin2[oi] = fminf(pmin[o][0], pmin[o][1]);
    atomicAdd(&sum2[o],   psum[o][0] + psum[o][1]);
    atomicAdd(&sumsq2[o], pss[o][0]  + pss[o][1]);
  }
}

// ---------------------------------------------------------------------------
// K9: final activation -> d_out
// ---------------------------------------------------------------------------
__global__ void k_out(const float* __restrict__ ymax2, const float* __restrict__ ymin2,
                      const float* __restrict__ scale2, const float* __restrict__ shift2,
                      float* __restrict__ out) {
  size_t i = (size_t)blockIdx.x * blockDim.x + threadIdx.x;
  if (i >= (size_t)BB * DD * NN) return;
  int o = (int)((i >> 12) & 63);
  float sc = scale2[o];
  float v  = sc * (sc >= 0.f ? ymax2[i] : ymin2[i]) + shift2[o];
  out[i] = (v >= 0.f) ? v : 0.2f * v;
}

// ---------------------------------------------------------------------------
extern "C" void kernel_launch(void* const* d_in, const int* in_sizes, int n_in,
                              void* d_out, int out_size, void* d_ws, size_t ws_size,
                              hipStream_t stream) {
  const float* x  = (const float*)d_in[0];
  const float* W1 = (const float*)d_in[1];
  const float* g1 = (const float*)d_in[2];
  const float* b1 = (const float*)d_in[3];
  const float* W2 = (const float*)d_in[4];
  const float* g2 = (const float*)d_in[5];
  const float* b2 = (const float*)d_in[6];
  float* out = (float*)d_out;

  // workspace carve-up (256B aligned)
  char* w = (char*)d_ws;
  size_t off = 0;
  auto take = [&](size_t bytes) -> void* {
    void* p = w + off;
    off = (off + bytes + 255) & ~(size_t)255;
    return p;
  };
  float*  stats  = (float*)take(512 * sizeof(float));
  float*  sum1   = stats,       *sumsq1 = stats + 64;
  float*  sum2   = stats + 128, *sumsq2 = stats + 192;
  float*  scale1 = stats + 256, *shift1 = stats + 320;
  float*  scale2 = stats + 384, *shift2 = stats + 448;
  float*  xx1   = (float*)take((size_t)BB * NN * sizeof(float));
  float*  xx2   = (float*)take((size_t)BB * NN * sizeof(float));
  int*    idx1  = (int*)  take((size_t)BB * NN * KK * sizeof(int));
  int*    idx2  = (int*)  take((size_t)BB * NN * KK * sizeof(int));
  float*  ymax1 = (float*)take((size_t)BB * DD * NN * sizeof(float));
  float*  ymin1 = (float*)take((size_t)BB * DD * NN * sizeof(float));
  float*  ymax2 = (float*)take((size_t)BB * DD * NN * sizeof(float));
  float*  ymin2 = (float*)take((size_t)BB * DD * NN * sizeof(float));
  float*  x1f   = (float*)take((size_t)BB * DD * NN * sizeof(float));
  __bf16* x1h   = (__bf16*)take((size_t)BB * DD * NN * sizeof(__bf16));

  const size_t elems = (size_t)BB * DD * NN;   // 2,097,152

  k_init_stats<<<1, 512, 0, stream>>>(stats);
  k_xx1<<<(BB * NN + 255) / 256, 256, 0, stream>>>(x, xx1);
  k_knn1<<<BB * (NN / 4), 128, 0, stream>>>(x, xx1, idx1);
  k_conv1<<<BB * (NN / 4), 256, 0, stream>>>(x, idx1, W1, ymax1, ymin1, sum1, sumsq1);
  k_finalize<<<1, 64, 0, stream>>>(sum1, sumsq1, g1, b1, scale1, shift1, CNT1);
  k_act1<<<(int)((elems + 255) / 256), 256, 0, stream>>>(ymax1, ymin1, scale1, shift1, x1f, x1h);
  k_xx2<<<(BB * NN + 255) / 256, 256, 0, stream>>>(x1f, xx2);
  k_knn2<<<BB * (NN / 16), 512, 0, stream>>>(x1h, xx2, idx2);
  k_conv2<<<BB * NN, 256, 0, stream>>>(x1f, W2, idx2, ymax2, ymin2, sum2, sumsq2);
  k_finalize<<<1, 64, 0, stream>>>(sum2, sumsq2, g2, b2, scale2, shift2, CNT1);
  k_out<<<(int)((elems + 255) / 256), 256, 0, stream>>>(ymax2, ymin2, scale2, shift2, out);
}
